// EncoderBlock_42485816492611
// MI455X (gfx1250) — matrix operand order repaired and checked by reading, not compiled, not run on hardware
//
#include <hip/hip_runtime.h>

typedef __attribute__((ext_vector_type(16))) __bf16 v16bf;
typedef __attribute__((ext_vector_type(8)))  float  v8f;
typedef __attribute__((ext_vector_type(4)))  int    v4i;
typedef __attribute__((ext_vector_type(8)))  int    v8i;

struct F8 { unsigned int x[8]; };

static __device__ __forceinline__ unsigned short f2bf(float f) {
  unsigned int u = __float_as_uint(f);
  u += 0x7FFFu + ((u >> 16) & 1u);   // round to nearest even
  return (unsigned short)(u >> 16);
}

static __device__ __forceinline__ v8f wmma_bf16(const F8& a, const F8& b, v8f c) {
  v16bf av = __builtin_bit_cast(v16bf, a);
  v16bf bv = __builtin_bit_cast(v16bf, b);
  return __builtin_amdgcn_wmma_f32_16x16x32_bf16(false, av, false, bv, (short)0, c, false, false);
}

// ---- Tensor Data Mover: DMA a 2-D tile (tile_d1 rows x tile_d0 elems of 2B)
// from global (row stride = row_stride_elems) into LDS at lds_off, inserting
// pad_amt dwords of LDS padding every pad_int dwords (D# §8.3/8.4 layout).
// pad codes: interval 2^(c+1) dwords; amount (c+1) dwords.
static __device__ __forceinline__ void tdm_load_2d(
    unsigned lds_off, const void* gptr, unsigned tile_d0, unsigned tile_d1,
    unsigned pad_int_code, unsigned pad_amt_code, unsigned row_stride_elems) {
  unsigned long long ga = (unsigned long long)(size_t)gptr;
  v4i g0;
  g0[0] = 1;                                                    // count=1, user mode
  g0[1] = (int)lds_off;                                         // lds_addr (bytes)
  g0[2] = (int)(unsigned)(ga & 0xFFFFFFFFu);                    // global_addr[31:0]
  g0[3] = (int)(((unsigned)(ga >> 32) & 0x01FFFFFFu) | (2u << 30)); // addr[56:32] | type=2
  unsigned td0 = row_stride_elems;      // tensor_dim0 (elements per line)
  unsigned td1 = 1u << 24;              // tensor_dim1: huge (tiles always in-bounds)
  unsigned long long st0 = row_stride_elems; // tensor_dim0_stride (elements)
  v8i g1;
  g1[0] = (int)((1u << 16) |            // data_size = 2 bytes
                (1u << 20) |            // pad_enable
                (pad_int_code << 22) | (pad_amt_code << 25));
  g1[1] = (int)((td0 & 0xFFFFu) << 16);                              // td0[15:0] -> bits 63:48
  g1[2] = (int)(((td0 >> 16) & 0xFFFFu) | ((td1 & 0xFFFFu) << 16));  // td0[31:16], td1[15:0]
  g1[3] = (int)(((td1 >> 16) & 0xFFFFu) | ((tile_d0 & 0xFFFFu) << 16)); // td1[31:16], tile_dim0
  g1[4] = (int)(tile_d1 & 0xFFFFu);                                  // tile_dim1 (tile_dim2=0)
  g1[5] = (int)(unsigned)(st0 & 0xFFFFFFFFu);                        // stride0[31:0]
  g1[6] = (int)(unsigned)((st0 >> 32) & 0xFFFFu);                    // stride0[47:32]
  g1[7] = 0;
  asm volatile("tensor_load_to_lds %0, %1" :: "s"(g0), "s"(g1) : "memory");
}

template<short N>
static __device__ __forceinline__ void wait_tensorcnt() {
  __builtin_amdgcn_s_wait_tensorcnt(N);
}

// ---------------- fp32 -> bf16 cast ----------------
__global__ void cast_bf16_kernel(const float* __restrict__ src,
                                 unsigned short* __restrict__ dst, long n) {
  long i = (long)blockIdx.x * blockDim.x + threadIdx.x;
  long stride = (long)gridDim.x * blockDim.x;
  for (; i < n; i += stride) dst[i] = f2bf(src[i]);
}

// ---------------- tiled WMMA GEMM: C[M,N] = A[M,K] * B[N,K]^T ----------------
// EPI: 0 = store bf16; 1 = store f32; 2 = bias + SiLU -> bf16; 3 = bias -> f32
#define TM 128
#define TN 128
#define TK 32
#define LSTR 40   // ushort stride = 80B/row: TDM pad (16dw interval + 4dw pad)

template<int EPI>
__global__ __launch_bounds__(256) void gemm_bf16_kernel(
    const unsigned short* __restrict__ A,
    const unsigned short* __restrict__ Bm,
    const float* __restrict__ bias,
    float* __restrict__ Cf,
    unsigned short* __restrict__ Cb,
    int M, int N, int Kd) {
  __shared__ __align__(16) unsigned short As[2][TM * LSTR];
  __shared__ __align__(16) unsigned short Bs[2][TN * LSTR];

  const int tid  = threadIdx.x;
  const int lane = tid & 31;
  const int wave = tid >> 5;
  const int wm   = wave >> 2;      // 0..1
  const int wn   = wave & 3;       // 0..3
  const int ln16 = lane & 15;
  const int half = lane >> 4;
  const int m0 = blockIdx.y * TM;
  const int n0 = blockIdx.x * TN;

  v8f acc[4][2];
  for (int i = 0; i < 4; i++)
    for (int j = 0; j < 2; j++)
      acc[i][j] = (v8f){0.f,0.f,0.f,0.f,0.f,0.f,0.f,0.f};

  const int nk = Kd / TK;
  if (wave == 0) {
    tdm_load_2d((unsigned)(size_t)&As[0][0], A  + (size_t)m0 * Kd, TK, TM, 3, 3, Kd);
    tdm_load_2d((unsigned)(size_t)&Bs[0][0], Bm + (size_t)n0 * Kd, TK, TN, 3, 3, Kd);
  }

  for (int i = 0; i < nk; i++) {
    const int cur = i & 1;
    if (wave == 0) {
      if (i + 1 < nk) {
        const int nxt = (i + 1) & 1;
        tdm_load_2d((unsigned)(size_t)&As[nxt][0],
                    A  + (size_t)m0 * Kd + (size_t)(i + 1) * TK, TK, TM, 3, 3, Kd);
        tdm_load_2d((unsigned)(size_t)&Bs[nxt][0],
                    Bm + (size_t)n0 * Kd + (size_t)(i + 1) * TK, TK, TN, 3, 3, Kd);
        wait_tensorcnt<2>();   // in-order: tile i's pair complete
      } else {
        wait_tensorcnt<0>();
      }
    }
    __syncthreads();   // tile i visible to all waves

    F8 af[4], bf[2];
    for (int ms = 0; ms < 4; ms++) {
      const unsigned short* base = &As[cur][(wm * 64 + ms * 16 + ln16) * LSTR];
      #pragma unroll
      for (int g = 0; g < 8; g++) {
        int kp = (g < 4) ? (half * 8 + 2 * g) : (16 + half * 8 + 2 * (g - 4));
        af[ms].x[g] = *(const unsigned int*)(base + kp);
      }
    }
    for (int ns = 0; ns < 2; ns++) {
      const unsigned short* base = &Bs[cur][(wn * 32 + ns * 16 + ln16) * LSTR];
      #pragma unroll
      for (int g = 0; g < 8; g++)
        bf[ns].x[g] = *(const unsigned int*)(base + ((g < 4) ? (half * 8 + 2 * g) : (16 + half * 8 + 2 * (g - 4))));
    }
    #pragma unroll
    for (int ms = 0; ms < 4; ms++)
      #pragma unroll
      for (int ns = 0; ns < 2; ns++)
        acc[ms][ns] = wmma_bf16(af[ms], bf[ns], acc[ms][ns]);

    __syncthreads();   // all reads of tile i done -> buffer reusable
  }

  for (int ms = 0; ms < 4; ms++) {
    for (int ns = 0; ns < 2; ns++) {
      int col = n0 + wn * 32 + ns * 16 + ln16;
      float bv = (EPI == 2 || EPI == 3) ? bias[col] : 0.f;
      #pragma unroll
      for (int r = 0; r < 8; r++) {
        int row = m0 + wm * 64 + ms * 16 + r + 8 * half;
        float v = acc[ms][ns][r];
        if (EPI == 2 || EPI == 3) v += bv;
        if (EPI == 2) v = v / (1.f + __expf(-v));  // SiLU
        if (EPI == 0 || EPI == 2) Cb[(size_t)row * N + col] = f2bf(v);
        else                      Cf[(size_t)row * N + col] = v;
      }
    }
  }
}

// ---------------- flash attention (mask == all ones) ----------------
// Q,K,V,O are bf16 [B*S, D] with head h in columns [h*64, h*64+64)
__global__ __launch_bounds__(128) void attn_kernel(
    const unsigned short* __restrict__ Q,
    const unsigned short* __restrict__ Kt,
    const unsigned short* __restrict__ V,
    unsigned short* __restrict__ O) {
  const int S = 2048, D = 1024, DH = 64;
  __shared__ __align__(16) unsigned short Ks[64 * 72];     // [key][d], TDM-padded rows
  __shared__ __align__(16) unsigned short Vs[64 * 72];     // [d][key] (transposed)
  __shared__ __align__(16) unsigned short Ps[4][16 * 72];  // per-wave P tile

  const int tid  = threadIdx.x;
  const int lane = tid & 31;
  const int wave = tid >> 5;
  const int ln16 = lane & 15;
  const int half = lane >> 4;
  const int bh = blockIdx.y;
  const int b  = bh >> 4;
  const int h  = bh & 15;
  const int q0 = blockIdx.x * 64 + wave * 16;
  const size_t rowbase = (size_t)b * S;
  const int colbase = h * DH;
  const unsigned ks_off = (unsigned)(size_t)&Ks[0];

  // Q fragments (A-layout), two K-steps over dh=64
  F8 qf[2];
  for (int ks = 0; ks < 2; ks++) {
    const unsigned short* qp = Q + (rowbase + q0 + ln16) * D + colbase + ks * 32;
    #pragma unroll
    for (int g = 0; g < 8; g++) {
      int kp = (g < 4) ? (half * 8 + 2 * g) : (16 + half * 8 + 2 * (g - 4));
      qf[ks].x[g] = *(const unsigned int*)(qp + kp);
    }
  }

  v8f o[4];
  for (int dt = 0; dt < 4; dt++) o[dt] = (v8f){0.f,0.f,0.f,0.f,0.f,0.f,0.f,0.f};
  float rm[8], rl[8];
  for (int r = 0; r < 8; r++) { rm[r] = -1e30f; rl[r] = 0.f; }

  const int ldr2 = tid >> 1;          // 0..63 (key row)
  const int ldc2 = (tid & 1) * 32;    // 0 or 32 (d column base)

  for (int kt = 0; kt < S / 64; kt++) {
    // K tile via TDM (64x64, row stride 72 ushorts via pad: 32dw interval + 4dw pad)
    if (wave == 0)
      tdm_load_2d(ks_off, Kt + (rowbase + (size_t)kt * 64) * D + colbase,
                  64, 64, 4, 3, D);
    // V tile transposed manually (overlaps with TDM)
    const unsigned short* vp = V + (rowbase + kt * 64 + ldr2) * D + colbase + ldc2;
    #pragma unroll
    for (int j = 0; j < 32; j++)
      Vs[(ldc2 + j) * 72 + ldr2] = vp[j];
    if (wave == 0) wait_tensorcnt<0>();
    __syncthreads();

    // scores: S = Q * K^T  (16 q-rows x 64 keys per wave)
    v8f sacc[4];
    for (int nt = 0; nt < 4; nt++) sacc[nt] = (v8f){0.f,0.f,0.f,0.f,0.f,0.f,0.f,0.f};
    #pragma unroll
    for (int ks = 0; ks < 2; ks++) {
      #pragma unroll
      for (int nt = 0; nt < 4; nt++) {
        F8 bf;
        const unsigned short* base = &Ks[(nt * 16 + ln16) * 72 + ks * 32];
        #pragma unroll
        for (int g = 0; g < 8; g++)
          bf.x[g] = *(const unsigned int*)(base + ((g < 4) ? (half * 8 + 2 * g) : (16 + half * 8 + 2 * (g - 4))));
        sacc[nt] = wmma_bf16(qf[ks], bf, sacc[nt]);
      }
    }

    // online softmax (scale 1/sqrt(64)=0.125); each row lives in 16 lanes of one half
    unsigned short* pw = &Ps[wave][0];
    #pragma unroll
    for (int r = 0; r < 8; r++) {
      float mx = -1e30f;
      for (int nt = 0; nt < 4; nt++) mx = fmaxf(mx, sacc[nt][r]);
      mx *= 0.125f;
      for (int off = 1; off < 16; off <<= 1) mx = fmaxf(mx, __shfl_xor(mx, off, 32));
      float mnew  = fmaxf(rm[r], mx);
      float alpha = __expf(rm[r] - mnew);
      float psum = 0.f;
      for (int nt = 0; nt < 4; nt++) {
        float pv = __expf(sacc[nt][r] * 0.125f - mnew);
        psum += pv;
        pw[(r + 8 * half) * 72 + nt * 16 + ln16] = f2bf(pv);
      }
      for (int off = 1; off < 16; off <<= 1) psum += __shfl_xor(psum, off, 32);
      rl[r] = rl[r] * alpha + psum;
      rm[r] = mnew;
      for (int dt = 0; dt < 4; dt++) o[dt][r] *= alpha;
    }

    // O += P * V  (A = P from wave-private LDS; B = transposed V tile)
    #pragma unroll
    for (int ks = 0; ks < 2; ks++) {
      F8 pf;
      #pragma unroll
      for (int g = 0; g < 8; g++) {
        int kp2 = (g < 4) ? (half * 8 + 2 * g) : (16 + half * 8 + 2 * (g - 4));
        pf.x[g] = *(const unsigned int*)&Ps[wave][ln16 * 72 + ks * 32 + kp2];
      }
      #pragma unroll
      for (int dt = 0; dt < 4; dt++) {
        F8 vf;
        const unsigned short* vb = &Vs[(dt * 16 + ln16) * 72 + ks * 32];
        #pragma unroll
        for (int g = 0; g < 8; g++)
          vf.x[g] = *(const unsigned int*)(vb + ((g < 4) ? (half * 8 + 2 * g) : (16 + half * 8 + 2 * (g - 4))));
        o[dt] = wmma_bf16(pf, vf, o[dt]);
      }
    }
    __syncthreads();
  }

  // normalize and write out
  for (int dt = 0; dt < 4; dt++) {
    #pragma unroll
    for (int r = 0; r < 8; r++) {
      float v = o[dt][r] / rl[r];
      int srow = q0 + r + 8 * half;
      O[(rowbase + srow) * D + colbase + dt * 16 + ln16] = f2bf(v);
    }
  }
}

// ---------------- fused residual add + LayerNorm ----------------
__global__ __launch_bounds__(256) void add_ln_kernel(
    const float* __restrict__ a, const float* __restrict__ b,
    const float* __restrict__ gamma, const float* __restrict__ beta,
    float* __restrict__ outf, unsigned short* outb) {
  const int D = 1024;
  __shared__ float ws1[8], ws2[8];
  const int row = blockIdx.x;
  const int tid = threadIdx.x;
  const int wave = tid >> 5, lane = tid & 31;

  float v[4], s = 0.f, s2 = 0.f;
  #pragma unroll
  for (int i = 0; i < 4; i++) {
    int c = tid + i * 256;
    float x = a[(size_t)row * D + c] + b[(size_t)row * D + c];
    v[i] = x; s += x; s2 += x * x;
  }
  for (int off = 1; off < 32; off <<= 1) {
    s  += __shfl_xor(s,  off, 32);
    s2 += __shfl_xor(s2, off, 32);
  }
  if (lane == 0) { ws1[wave] = s; ws2[wave] = s2; }
  __syncthreads();
  if (wave == 0) {
    float t1 = (lane < 8) ? ws1[lane] : 0.f;
    float t2 = (lane < 8) ? ws2[lane] : 0.f;
    for (int off = 1; off < 8; off <<= 1) {
      t1 += __shfl_xor(t1, off, 32);
      t2 += __shfl_xor(t2, off, 32);
    }
    if (lane == 0) { ws1[0] = t1; ws2[0] = t2; }
  }
  __syncthreads();
  float mean = ws1[0] * (1.f / 1024.f);
  float var  = ws2[0] * (1.f / 1024.f) - mean * mean;
  float rstd = rsqrtf(var + 1e-6f);
  #pragma unroll
  for (int i = 0; i < 4; i++) {
    int c = tid + i * 256;
    float z = (v[i] - mean) * rstd * gamma[c] + beta[c];
    outf[(size_t)row * D + c] = z;
    if (outb) outb[(size_t)row * D + c] = f2bf(z);
  }
}

// ---------------- host orchestration ----------------
extern "C" void kernel_launch(void* const* d_in, const int* in_sizes, int n_in,
                              void* d_out, int out_size, void* d_ws, size_t ws_size,
                              hipStream_t stream) {
  (void)in_sizes; (void)n_in; (void)out_size; (void)ws_size;
  const int B = 4, S = 2048, D = 1024, FF = 4096;
  const int M = B * S;  // 8192

  const float* x   = (const float*)d_in[0];
  const float* W_Q = (const float*)d_in[2];
  const float* W_K = (const float*)d_in[3];
  const float* W_V = (const float*)d_in[4];
  const float* W_O = (const float*)d_in[5];
  const float* W1  = (const float*)d_in[6];
  const float* b1  = (const float*)d_in[7];
  const float* W2  = (const float*)d_in[8];
  const float* b2  = (const float*)d_in[9];
  const float* g1  = (const float*)d_in[10];
  const float* be1 = (const float*)d_in[11];
  const float* g2  = (const float*)d_in[12];
  const float* be2 = (const float*)d_in[13];
  float* out = (float*)d_out;

  char* ws = (char*)d_ws;
  size_t off = 0;
  auto alloc = [&](size_t bytes) { char* p = ws + off; off += (bytes + 255) & ~(size_t)255; return p; };
  unsigned short* xb  = (unsigned short*)alloc((size_t)M * D * 2);
  unsigned short* wqb = (unsigned short*)alloc((size_t)D * D * 2);
  unsigned short* wkb = (unsigned short*)alloc((size_t)D * D * 2);
  unsigned short* wvb = (unsigned short*)alloc((size_t)D * D * 2);
  unsigned short* wob = (unsigned short*)alloc((size_t)D * D * 2);
  unsigned short* w1b = (unsigned short*)alloc((size_t)FF * D * 2);
  unsigned short* w2b = (unsigned short*)alloc((size_t)D * FF * 2);
  unsigned short* qb  = (unsigned short*)alloc((size_t)M * D * 2);
  unsigned short* kb  = (unsigned short*)alloc((size_t)M * D * 2);
  unsigned short* vb  = (unsigned short*)alloc((size_t)M * D * 2);
  unsigned short* ab  = (unsigned short*)alloc((size_t)M * D * 2);
  float*          t1f = (float*)alloc((size_t)M * D * 4);   // also reused for ff
  float*          t4f = (float*)alloc((size_t)M * D * 4);
  unsigned short* t4b = (unsigned short*)alloc((size_t)M * D * 2);
  unsigned short* h1b = (unsigned short*)alloc((size_t)M * FF * 2);

  auto cast = [&](const float* s, unsigned short* dst, long n) {
    cast_bf16_kernel<<<4096, 256, 0, stream>>>(s, dst, n);
  };
  cast(x,   xb,  (long)M * D);
  cast(W_Q, wqb, (long)D * D);
  cast(W_K, wkb, (long)D * D);
  cast(W_V, wvb, (long)D * D);
  cast(W1,  w1b, (long)FF * D);
  cast(W_O, wob, (long)D * D);
  cast(W2,  w2b, (long)D * FF);

  dim3 blk(256);
  dim3 gDD(D / TN, M / TM);    // N=1024
  dim3 gFF(FF / TN, M / TM);   // N=4096

  // Q, K, V projections -> bf16
  gemm_bf16_kernel<0><<<gDD, blk, 0, stream>>>(xb, wqb, nullptr, nullptr, qb, M, D, D);
  gemm_bf16_kernel<0><<<gDD, blk, 0, stream>>>(xb, wkb, nullptr, nullptr, kb, M, D, D);
  gemm_bf16_kernel<0><<<gDD, blk, 0, stream>>>(xb, wvb, nullptr, nullptr, vb, M, D, D);

  // flash attention -> concat head outputs (bf16)
  attn_kernel<<<dim3(S / 64, B * 16), dim3(128), 0, stream>>>(qb, kb, vb, ab);

  // output projection -> f32
  gemm_bf16_kernel<1><<<gDD, blk, 0, stream>>>(ab, wob, nullptr, t1f, nullptr, M, D, D);

  // t4 = LN(t1 + x)  (f32 + bf16 copy)
  add_ln_kernel<<<M, blk, 0, stream>>>(t1f, x, g1, be1, t4f, t4b);

  // h1 = silu(t4 @ W1^T + b1) -> bf16
  gemm_bf16_kernel<2><<<gFF, blk, 0, stream>>>(t4b, w1b, b1, nullptr, h1b, M, FF, D);

  // ff = h1 @ W2^T + b2 -> f32 (reuse t1f)
  gemm_bf16_kernel<3><<<gDD, blk, 0, stream>>>(h1b, w2b, b2, t1f, nullptr, M, D, FF);

  // out = LN(ff + t4)
  add_ln_kernel<<<M, blk, 0, stream>>>(t1f, t4f, g2, be2, out, nullptr);
}
